// SparseMoEMLP_17343077941500
// MI455X (gfx1250) — matrix operand order repaired and checked by reading, not compiled, not run on hardware
//
#include <hip/hip_runtime.h>
#include <math.h>

// Problem constants (B=2, S=1024, H=1024, I=2048, E=8, TOP_K=2)
#define TOKENS 2048
#define HDIM   1024
#define IDIM   2048
#define NEXP   8

typedef __attribute__((ext_vector_type(16))) __bf16        v16bf;
typedef __attribute__((ext_vector_type(8)))  float         v8f;
typedef __attribute__((ext_vector_type(4)))  float         v4f;
typedef __attribute__((ext_vector_type(4)))  unsigned int  v4u;
typedef __attribute__((ext_vector_type(4)))  int           v4i;

// Address-space-qualified pointee types for the async-LDS builtin.
typedef __attribute__((address_space(1))) v4i as1_v4i;
typedef __attribute__((address_space(3))) v4i as3_v4i;

union Frag16 {
    v16bf          bf;
    v4u            q[2];
    unsigned short us[16];
};

__device__ __forceinline__ unsigned short f2bf(float f) {
    return __builtin_bit_cast(unsigned short, (__bf16)f);   // native v_cvt, RNE
}

__device__ __forceinline__ float gelu_exact(float x) {
    return 0.5f * x * (1.0f + erff(x * 0.70710678118654752440f));
}

__device__ __forceinline__ v8f zero8() {
    v8f z;
#pragma unroll
    for (int i = 0; i < 8; ++i) z[i] = 0.0f;
    return z;
}

__device__ __forceinline__ v8f wmma_bf16(const Frag16& a, const Frag16& b, v8f c) {
    return __builtin_amdgcn_wmma_f32_16x16x32_bf16(
        /*neg_a=*/false, a.bf, /*neg_b=*/false, b.bf,
        /*c_mod=*/(short)0, c, /*reuse_a=*/false, /*reuse_b=*/false);
}

// ---- gfx1250 async global->LDS copy (16B per lane), with safe fallback ----
#define HAVE_ASYNC_LDS \
    (__has_builtin(__builtin_amdgcn_global_load_async_to_lds_b128))

__device__ __forceinline__ void async_copy16(const unsigned short* g, unsigned short* l) {
#if HAVE_ASYNC_LDS
    __builtin_amdgcn_global_load_async_to_lds_b128(
        (as1_v4i*)(g), (as3_v4i*)(l), 0, 0);
#else
    *(v4u*)l = *(const v4u*)g;
#endif
}

__device__ __forceinline__ void async_wait_all() {
#if HAVE_ASYNC_LDS
  #if __has_builtin(__builtin_amdgcn_s_wait_asynccnt)
    __builtin_amdgcn_s_wait_asynccnt(0);
  #else
    asm volatile("s_wait_asynccnt 0x0" ::: "memory");
  #endif
#endif
}

// B-fragment loader from global f32 with native convert (fallback path).
__device__ __forceinline__ void load_b_frag_f32(Frag16& b, const void* base, size_t off) {
    const float* p = (const float*)base + off;
#pragma unroll
    for (int j = 0; j < 16; j += 4) {
        v4f f = *(const v4f*)(p + j);
#pragma unroll
        for (int q = 0; q < 4; ++q) b.bf[j + q] = (__bf16)f[q];
    }
}

// ---------------------------------------------------------------------------
// Zero-init the MoE output region of d_out (harness poisons it).
// ---------------------------------------------------------------------------
__global__ void moe_zero_f32(float* __restrict__ p, int n) {
    int i = blockIdx.x * blockDim.x + threadIdx.x;
    if (i < n) p[i] = 0.0f;
}

// ---------------------------------------------------------------------------
// Convert f32 -> bf16, 4 elements/thread (x and, ws permitting, weights).
// ---------------------------------------------------------------------------
__global__ void moe_cvt_bf16(const float* __restrict__ in,
                             unsigned short* __restrict__ outp, int n4) {
    int i = blockIdx.x * blockDim.x + threadIdx.x;
    if (i < n4) {
        v4f f = ((const v4f*)in)[i];
        union { unsigned short us[4]; unsigned long long u; } pk;
#pragma unroll
        for (int q = 0; q < 4; ++q) pk.us[q] = f2bf(f[q]);
        ((unsigned long long*)outp)[i] = pk.u;
    }
}

// ---------------------------------------------------------------------------
// Router: logits = x @ rw^T (full f32), softmax, top-2, renormalized gates.
// ---------------------------------------------------------------------------
__global__ __launch_bounds__(256)
void moe_router(const float* __restrict__ x, const float* __restrict__ rw,
                float* __restrict__ logits_out, float* __restrict__ gates) {
    const int lane = threadIdx.x & 31;
    const int wave = threadIdx.x >> 5;
    const int t = blockIdx.x * 8 + wave;
    if (t >= TOKENS) return;

    float acc[NEXP];
#pragma unroll
    for (int e = 0; e < NEXP; ++e) acc[e] = 0.0f;

    for (int h = lane; h < HDIM; h += 32) {
        float xv = x[(size_t)t * HDIM + h];
#pragma unroll
        for (int e = 0; e < NEXP; ++e)
            acc[e] = fmaf(xv, rw[(size_t)e * HDIM + h], acc[e]);
    }
#pragma unroll
    for (int e = 0; e < NEXP; ++e) {
#pragma unroll
        for (int m = 16; m >= 1; m >>= 1)
            acc[e] += __shfl_xor(acc[e], m, 32);
    }
    if (lane == 0) {
        float mx = acc[0];
#pragma unroll
        for (int e = 1; e < NEXP; ++e) mx = fmaxf(mx, acc[e]);
        float p[NEXP];
#pragma unroll
        for (int e = 0; e < NEXP; ++e) p[e] = __expf(acc[e] - mx);
        int i0 = 0;
#pragma unroll
        for (int e = 1; e < NEXP; ++e) if (p[e] > p[i0]) i0 = e;
        int i1 = (i0 == 0) ? 1 : 0;
#pragma unroll
        for (int e = 0; e < NEXP; ++e)
            if (e != i0 && p[e] > p[i1]) i1 = e;
        float denom = p[i0] + p[i1];
#pragma unroll
        for (int e = 0; e < NEXP; ++e) {
            float g = (e == i0) ? (p[i0] / denom)
                    : (e == i1) ? (p[i1] / denom) : 0.0f;
            gates[(size_t)t * NEXP + e] = g;
            logits_out[(size_t)t * NEXP + e] = acc[e];
        }
    }
}

// ---------------------------------------------------------------------------
// GEMM1: h[T,I] = gelu(x @ w1^T) * (x @ v^T) for one expert, bf16 output.
// 128 threads = 4 waves; block tile 64x64; wave tile 32x32; K step 32.
// BF path: B tiles staged in LDS via async copies shared by all 4 waves.
// ---------------------------------------------------------------------------
template <bool BF>
__global__ __launch_bounds__(128)
void moe_gemm1(const unsigned short* __restrict__ xb,  // [T,H] bf16
               const void* __restrict__ w1e,           // [I,H] expert slice
               const void* __restrict__ ve,            // [I,H] expert slice
               unsigned short* __restrict__ hout) {    // [T,I] bf16
    const int lane = threadIdx.x & 31;
    const int wave = threadIdx.x >> 5;
    const int wm = wave >> 1, wn = wave & 1;
    const int Mbase = blockIdx.y * 64 + wm * 32;
    const int Nbase = blockIdx.x * 64 + wn * 32;
    const int lr  = lane & 15;
    const int lhi = lane >> 4;

    v8f acc1[2][2], acc2[2][2];
#pragma unroll
    for (int i = 0; i < 2; ++i)
#pragma unroll
        for (int j = 0; j < 2; ++j) { acc1[i][j] = zero8(); acc2[i][j] = zero8(); }

    if constexpr (BF) {
        __shared__ unsigned short sB1[64 * 32];   // [n_local][k] bf16, 4 KB
        __shared__ unsigned short sB2[64 * 32];

        // Staging assignment: lane covers row srow, element chunks lhi*8, lhi*8+16.
        const int srow = wave * 16 + lr;                         // 0..63
        const unsigned short* gB1 =
            (const unsigned short*)w1e + (size_t)(blockIdx.x * 64 + srow) * HDIM;
        const unsigned short* gB2 =
            (const unsigned short*)ve + (size_t)(blockIdx.x * 64 + srow) * HDIM;
        unsigned short* lB1 = sB1 + srow * 32;
        unsigned short* lB2 = sB2 + srow * 32;
        const int c0 = lhi * 8, c1 = lhi * 8 + 16;

        // Prologue: issue k = 0 tile.
        async_copy16(gB1 + c0, lB1 + c0);
        async_copy16(gB1 + c1, lB1 + c1);
        async_copy16(gB2 + c0, lB2 + c0);
        async_copy16(gB2 + c1, lB2 + c1);

        for (int k = 0; k < HDIM; k += 32) {
            async_wait_all();
            __syncthreads();
            Frag16 b1[2], b2[2];
#pragma unroll
            for (int ni = 0; ni < 2; ++ni) {
                const int rl = (wn * 32 + ni * 16 + lr) * 32 + lhi * 8;
                b1[ni].q[0] = *(const v4u*)(sB1 + rl);
                b1[ni].q[1] = *(const v4u*)(sB1 + rl + 16);
                b2[ni].q[0] = *(const v4u*)(sB2 + rl);
                b2[ni].q[1] = *(const v4u*)(sB2 + rl + 16);
            }
            __syncthreads();
            if (k + 32 < HDIM) {   // uniform guard; overlaps with WMMAs below
                async_copy16(gB1 + k + 32 + c0, lB1 + c0);
                async_copy16(gB1 + k + 32 + c1, lB1 + c1);
                async_copy16(gB2 + k + 32 + c0, lB2 + c0);
                async_copy16(gB2 + k + 32 + c1, lB2 + c1);
            }
            Frag16 a[2];
#pragma unroll
            for (int mi = 0; mi < 2; ++mi) {
                const unsigned short* p =
                    xb + (size_t)(Mbase + mi * 16 + lr) * HDIM + k + lhi * 8;
                a[mi].q[0] = *(const v4u*)(p);
                a[mi].q[1] = *(const v4u*)(p + 16);
                __builtin_prefetch(p + 32, 0, 3);
            }
#pragma unroll
            for (int mi = 0; mi < 2; ++mi)
#pragma unroll
                for (int ni = 0; ni < 2; ++ni) {
                    acc1[mi][ni] = wmma_bf16(a[mi], b1[ni], acc1[mi][ni]);
                    acc2[mi][ni] = wmma_bf16(a[mi], b2[ni], acc2[mi][ni]);
                }
        }
    } else {
        for (int k = 0; k < HDIM; k += 32) {
            Frag16 a[2];
#pragma unroll
            for (int mi = 0; mi < 2; ++mi) {
                const unsigned short* p =
                    xb + (size_t)(Mbase + mi * 16 + lr) * HDIM + k + lhi * 16;
                a[mi].q[0] = *(const v4u*)(p);
                a[mi].q[1] = *(const v4u*)(p + 8);
            }
            Frag16 b1[2], b2[2];
#pragma unroll
            for (int ni = 0; ni < 2; ++ni) {
                const size_t roff = (size_t)(Nbase + ni * 16 + lr) * HDIM + k + lhi * 16;
                load_b_frag_f32(b1[ni], w1e, roff);
                load_b_frag_f32(b2[ni], ve, roff);
            }
#pragma unroll
            for (int mi = 0; mi < 2; ++mi)
#pragma unroll
                for (int ni = 0; ni < 2; ++ni) {
                    acc1[mi][ni] = wmma_bf16(a[mi], b1[ni], acc1[mi][ni]);
                    acc2[mi][ni] = wmma_bf16(a[mi], b2[ni], acc2[mi][ni]);
                }
        }
    }

#pragma unroll
    for (int mi = 0; mi < 2; ++mi)
#pragma unroll
        for (int ni = 0; ni < 2; ++ni)
#pragma unroll
            for (int r = 0; r < 8; ++r) {
                int row = Mbase + mi * 16 + lhi * 8 + r;
                int col = Nbase + ni * 16 + lr;
                float hv = gelu_exact(acc1[mi][ni][r]) * acc2[mi][ni][r];
                hout[(size_t)row * IDIM + col] = f2bf(hv);
            }
}

// ---------------------------------------------------------------------------
// GEMM2: out[T,H] += gate_e(t) * (h @ w2^T) for one expert.
// ---------------------------------------------------------------------------
template <bool BF>
__global__ __launch_bounds__(128)
void moe_gemm2(const unsigned short* __restrict__ hb,  // [T,I] bf16
               const void* __restrict__ w2e,           // [H,I] expert slice
               const float* __restrict__ gates,        // [T,E]
               int e,
               float* __restrict__ out) {              // [T,H] f32, accumulated
    const int lane = threadIdx.x & 31;
    const int wave = threadIdx.x >> 5;
    const int wm = wave >> 1, wn = wave & 1;
    const int Mbase = blockIdx.y * 64 + wm * 32;
    const int Nbase = blockIdx.x * 64 + wn * 32;
    const int lr  = lane & 15;
    const int lhi = lane >> 4;

    v8f acc[2][2];
#pragma unroll
    for (int i = 0; i < 2; ++i)
#pragma unroll
        for (int j = 0; j < 2; ++j) acc[i][j] = zero8();

    if constexpr (BF) {
        __shared__ unsigned short sB[64 * 32];   // 4 KB

        const int srow = wave * 16 + lr;
        const unsigned short* gB =
            (const unsigned short*)w2e + (size_t)(blockIdx.x * 64 + srow) * IDIM;
        unsigned short* lB = sB + srow * 32;
        const int c0 = lhi * 8, c1 = lhi * 8 + 16;

        async_copy16(gB + c0, lB + c0);
        async_copy16(gB + c1, lB + c1);

        for (int k = 0; k < IDIM; k += 32) {
            async_wait_all();
            __syncthreads();
            Frag16 b[2];
#pragma unroll
            for (int ni = 0; ni < 2; ++ni) {
                const int rl = (wn * 32 + ni * 16 + lr) * 32 + lhi * 8;
                b[ni].q[0] = *(const v4u*)(sB + rl);
                b[ni].q[1] = *(const v4u*)(sB + rl + 16);
            }
            __syncthreads();
            if (k + 32 < IDIM) {
                async_copy16(gB + k + 32 + c0, lB + c0);
                async_copy16(gB + k + 32 + c1, lB + c1);
            }
            Frag16 a[2];
#pragma unroll
            for (int mi = 0; mi < 2; ++mi) {
                const unsigned short* p =
                    hb + (size_t)(Mbase + mi * 16 + lr) * IDIM + k + lhi * 8;
                a[mi].q[0] = *(const v4u*)(p);
                a[mi].q[1] = *(const v4u*)(p + 16);
                __builtin_prefetch(p + 32, 0, 3);
            }
#pragma unroll
            for (int mi = 0; mi < 2; ++mi)
#pragma unroll
                for (int ni = 0; ni < 2; ++ni)
                    acc[mi][ni] = wmma_bf16(a[mi], b[ni], acc[mi][ni]);
        }
    } else {
        for (int k = 0; k < IDIM; k += 32) {
            Frag16 a[2];
#pragma unroll
            for (int mi = 0; mi < 2; ++mi) {
                const unsigned short* p =
                    hb + (size_t)(Mbase + mi * 16 + lr) * IDIM + k + lhi * 16;
                a[mi].q[0] = *(const v4u*)(p);
                a[mi].q[1] = *(const v4u*)(p + 8);
            }
            Frag16 b[2];
#pragma unroll
            for (int ni = 0; ni < 2; ++ni) {
                const size_t roff = (size_t)(Nbase + ni * 16 + lr) * IDIM + k + lhi * 16;
                load_b_frag_f32(b[ni], w2e, roff);
            }
#pragma unroll
            for (int mi = 0; mi < 2; ++mi)
#pragma unroll
                for (int ni = 0; ni < 2; ++ni)
                    acc[mi][ni] = wmma_bf16(a[mi], b[ni], acc[mi][ni]);
        }
    }

#pragma unroll
    for (int mi = 0; mi < 2; ++mi)
#pragma unroll
        for (int ni = 0; ni < 2; ++ni)
#pragma unroll
            for (int r = 0; r < 8; ++r) {
                int row = Mbase + mi * 16 + lhi * 8 + r;
                int col = Nbase + ni * 16 + lr;
                float g = gates[(size_t)row * NEXP + e];
                out[(size_t)row * HDIM + col] += g * acc[mi][ni][r];
            }
}

// ---------------------------------------------------------------------------
// Launch
// ---------------------------------------------------------------------------
extern "C" void kernel_launch(void* const* d_in, const int* in_sizes, int n_in,
                              void* d_out, int out_size, void* d_ws, size_t ws_size,
                              hipStream_t stream) {
    (void)in_sizes; (void)n_in; (void)out_size;

    const float* x   = (const float*)d_in[0];   // [B,S,H]
    const float* rw  = (const float*)d_in[1];   // [E,H]
    const float* w1  = (const float*)d_in[2];   // [E,I,H]
    const float* v   = (const float*)d_in[3];   // [E,I,H]
    const float* w2  = (const float*)d_in[4];   // [E,H,I]

    float* out    = (float*)d_out;                       // [T,H]
    float* logits = out + (size_t)TOKENS * HDIM;         // [T,E]

    char* ws = (char*)d_ws;
    size_t off = 0;
    unsigned short* xb = (unsigned short*)(ws + off); off += (size_t)TOKENS * HDIM * 2;
    unsigned short* hb = (unsigned short*)(ws + off); off += (size_t)TOKENS * IDIM * 2;
    float* gates       = (float*)(ws + off);          off += (size_t)TOKENS * NEXP * 4;

    const size_t wElems = (size_t)NEXP * IDIM * HDIM;   // per weight tensor
    const bool big = ws_size >= off + 3 * wElems * 2;   // room for bf16 weights?
    unsigned short* w1b = nullptr; unsigned short* vb = nullptr; unsigned short* w2b = nullptr;
    if (big) {
        w1b = (unsigned short*)(ws + off); off += wElems * 2;
        vb  = (unsigned short*)(ws + off); off += wElems * 2;
        w2b = (unsigned short*)(ws + off); off += wElems * 2;
    }

    const int nOut = TOKENS * HDIM;
    moe_zero_f32<<<(nOut + 255) / 256, 256, 0, stream>>>(out, nOut);

    const int n4 = (TOKENS * HDIM) / 4;
    moe_cvt_bf16<<<(n4 + 255) / 256, 256, 0, stream>>>(x, xb, n4);

    moe_router<<<TOKENS / 8, 256, 0, stream>>>(x, rw, logits, gates);

    if (big) {
        const int w4 = (int)(wElems / 4);
        moe_cvt_bf16<<<(w4 + 255) / 256, 256, 0, stream>>>(w1, w1b, w4);
        moe_cvt_bf16<<<(w4 + 255) / 256, 256, 0, stream>>>(v,  vb,  w4);
        moe_cvt_bf16<<<(w4 + 255) / 256, 256, 0, stream>>>(w2, w2b, w4);
    }

    const size_t wsz = (size_t)IDIM * HDIM;
    for (int e = 0; e < NEXP; ++e) {
        if (big) {
            moe_gemm1<true><<<dim3(IDIM / 64, TOKENS / 64), 128, 0, stream>>>(
                xb, w1b + e * wsz, vb + e * wsz, hb);
            moe_gemm2<true><<<dim3(HDIM / 64, TOKENS / 64), 128, 0, stream>>>(
                hb, w2b + e * wsz, gates, e, out);
        } else {
            moe_gemm1<false><<<dim3(IDIM / 64, TOKENS / 64), 128, 0, stream>>>(
                xb, w1 + e * wsz, v + e * wsz, hb);
            moe_gemm2<false><<<dim3(HDIM / 64, TOKENS / 64), 128, 0, stream>>>(
                hb, w2 + e * wsz, gates, e, out);
        }
    }
}
